// QLSTM_65481071402108
// MI455X (gfx1250) — matrix operand order repaired and checked by reading, not compiled, not run on hardware
//
#include <hip/hip_runtime.h>
#include <stdint.h>

// ---------------------------------------------------------------------------
// QLSTM on MI455X (gfx1250): bf16 WMMA GEMMs, 2 M-tiles x 4 gates per wave
// (register blocking halves L2 weight traffic), fused quantum path whose
// per-kb partials are anchored in LDS (prevents VALU sinking into the WMMA
// region -> spill-free), LDS-staged A, prefetch pipelining.
// ---------------------------------------------------------------------------
#define T_STEPS 256
#define BATCH   512
#define DIM     512
#define HID     512
#define INW     1024
#define NQ      8
#define ALPHA   0.7f

typedef __attribute__((ext_vector_type(16))) __bf16 v16bf;
typedef __attribute__((ext_vector_type(8)))  float  v8f;

union FragAB { v16bf v; uint4 q[2]; };

__device__ __forceinline__ unsigned short f32_to_bf16_bits(float f) {
  unsigned u = __builtin_bit_cast(unsigned, f);
  u += 0x7FFFu + ((u >> 16) & 1u);           // round-to-nearest-even
  return (unsigned short)(u >> 16);
}
__device__ __forceinline__ __bf16 f32_to_bf16(float f) {
  unsigned short b = f32_to_bf16_bits(f);
  return __builtin_bit_cast(__bf16, b);
}
__device__ __forceinline__ unsigned pack_bf16x2(float lo, float hi) {
  return ((unsigned)f32_to_bf16_bits(hi) << 16) | (unsigned)f32_to_bf16_bits(lo);
}
__device__ __forceinline__ float sigmf(float x) { return 1.0f / (1.0f + __expf(-x)); }

// --- one-time weight prep: W (INW x HID, f32) -> WT[g][n][k] bf16 -----------
__global__ void qlstm_wprep(const float* __restrict__ Wf, const float* __restrict__ Wi,
                            const float* __restrict__ Wu, const float* __restrict__ Wo,
                            __bf16* __restrict__ WT) {
  unsigned idx = blockIdx.x * blockDim.x + threadIdx.x;   // 4*512*1024 threads
  unsigned g   = idx >> 19;
  unsigned rem = idx & 0x7FFFFu;
  unsigned n   = rem >> 10;
  unsigned k   = rem & 1023u;
  const float* W = (g == 0) ? Wf : (g == 1) ? Wi : (g == 2) ? Wu : Wo;
  WT[idx] = f32_to_bf16(W[(size_t)k * HID + n]);
}

// --- one-time quantum weight prep: Wq (INW x NQ) -> WqT[j][k] f32 -----------
__global__ void qlstm_wqprep(const float* __restrict__ Wq, float* __restrict__ WqT) {
  unsigned idx = blockIdx.x * blockDim.x + threadIdx.x;   // 8*1024 threads
  unsigned j = idx >> 10, k = idx & 1023u;
  WqT[idx] = Wq[(size_t)k * NQ + j];
}

__global__ void qlstm_zero(float* __restrict__ p, int n) {
  int i = blockIdx.x * blockDim.x + threadIdx.x;
  if (i < n) p[i] = 0.0f;
}

// --- main step: fused 4-gate WMMA GEMM (2 M-tiles/wave) + quantum path ------
// grid (BATCH/32, HID/64), block 128 (4 waves); wave -> 32x16 tile, 4 gates
__global__ __launch_bounds__(128, 2) void qlstm_step(
    const float* __restrict__ xt, const float* __restrict__ hx,
    const __bf16* __restrict__ WT,                   // [4][HID][INW] bf16
    const float* __restrict__ WqT,                   // [NQ][INW]     f32
    const float* __restrict__ bF, const float* __restrict__ bI,
    const float* __restrict__ bU, const float* __restrict__ bO,
    const float* __restrict__ Wqh, const float* __restrict__ bqh,
    const float* __restrict__ bq,
    const float* __restrict__ thf, const float* __restrict__ thi,
    const float* __restrict__ thu, const float* __restrict__ tho,
    float* __restrict__ cx, float* __restrict__ out_t) {
  __shared__ unsigned ldsA[32 * 128];          // 32 rows x 256 bf16 (16 KB)
  __shared__ float    qpartS[32 * 4 * NQ];     // partial q per (row, seg, j) (4 KB)
  __shared__ float    qinS[32 * NQ];           // q_in per block row (1 KB)
  __shared__ float    qcpS[4 * 32 * NQ];       // cumprod-cos per gate/row (4 KB)

  const int tid  = threadIdx.x;
  const int nt   = tid >> 5;                   // wave id = N-tile (0..3)
  const int lane = tid & 31;
  const int half = lane >> 4, lm = lane & 15;
  const int rowBase = blockIdx.x * 32;
  const int colBase = blockIdx.y * 64;
  const int h = colBase + nt * 16 + lm;

  v8f aF[2] = {{}, {}}, aI[2] = {{}, {}}, aU[2] = {{}, {}}, aO[2] = {{}, {}};

  const int srow = tid >> 2;          // 0..31 : staged row / q row
  const int seg  = tid & 3;           // 0..3  : 64-wide column segment
  const int scol = seg * 64;          // 0..192: staged column offset
  const int qslot = (srow * 4 + seg) * NQ;

  const __bf16* wpF = WT + (size_t)0 * HID * INW + (size_t)h * INW;
  const __bf16* wpI = WT + (size_t)1 * HID * INW + (size_t)h * INW;
  const __bf16* wpU = WT + (size_t)2 * HID * INW + (size_t)h * INW;
  const __bf16* wpO = WT + (size_t)3 * HID * INW + (size_t)h * INW;

#pragma unroll 1
  for (int kb = 0; kb < 4; ++kb) {
    // stage: load f32 source (two 32-float passes), pack bf16 -> LDS,
    // and compute this kb's quantum-path partial sums
    float qk[NQ];
#pragma unroll
    for (int j = 0; j < NQ; ++j) qk[j] = 0.0f;
#pragma unroll
    for (int hs = 0; hs < 2; ++hs) {
      const int col0 = scol + hs * 32;
      const float4* s4 = (kb < 2)
          ? (const float4*)(xt + (size_t)(rowBase + srow) * DIM + kb * 256 + col0)
          : (const float4*)(hx + (size_t)(rowBase + srow) * HID + (kb - 2) * 256 + col0);
      float4 c4[8];
#pragma unroll
      for (int i = 0; i < 8; ++i) c4[i] = s4[i];

      unsigned* dst = &ldsA[srow * 128 + (col0 >> 1)];
#pragma unroll
      for (int i = 0; i < 8; ++i) {
        dst[2 * i]     = pack_bf16x2(c4[i].x, c4[i].y);
        dst[2 * i + 1] = pack_bf16x2(c4[i].z, c4[i].w);
      }

      const float* wq0 = WqT + kb * 256 + col0;
#pragma unroll
      for (int j = 0; j < NQ; ++j) {
        const float4* w4 = (const float4*)(wq0 + (size_t)j * INW);
        float s = 0.0f;
#pragma unroll
        for (int i = 0; i < 8; ++i) {
          const float4 w = w4[i], c = c4[i];
          s += c.x * w.x + c.y * w.y + c.z * w.z + c.w * w.w;
        }
        qk[j] += s;
      }
    }
    // anchor the q FMA chains with LDS read-modify-writes: these ds ops
    // cannot cross the barrier, so all staging transients die here.
    if (kb == 0) {
#pragma unroll
      for (int j = 0; j < NQ; ++j) qpartS[qslot + j] = qk[j];
    } else {
#pragma unroll
      for (int j = 0; j < NQ; ++j) qpartS[qslot + j] += qk[j];
    }
    __syncthreads();

    // prefetch next kb's source segment + weight K-blocks (global_prefetch_b8)
    if (kb < 3) {
      const int kn = kb + 1;
      const char* nsrc = (kn < 2)
          ? (const char*)(xt + (size_t)(rowBase + srow) * DIM + kn * 256 + scol)
          : (const char*)(hx + (size_t)(rowBase + srow) * HID + (kn - 2) * 256 + scol);
      __builtin_prefetch(nsrc, 0, 3);
      __builtin_prefetch(nsrc + 128, 0, 3);
      const int kg_next = kn * 256;
      __builtin_prefetch(wpF + kg_next, 0, 3);
      __builtin_prefetch(wpI + kg_next, 0, 3);
      __builtin_prefetch(wpU + kg_next, 0, 3);
      __builtin_prefetch(wpO + kg_next, 0, 3);
    }

#pragma unroll
    for (int kl = 0; kl < 256; kl += 32) {
      const int kg = kb * 256 + kl + 16 * half;
      // all 4 gate B-fragments up-front: one big load clause, 8 WMMAs of cover
      FragAB fbF, fbI, fbU, fbO;
      { const uint4* p = (const uint4*)(wpF + kg); fbF.q[0] = p[0]; fbF.q[1] = p[1]; }
      { const uint4* p = (const uint4*)(wpI + kg); fbI.q[0] = p[0]; fbI.q[1] = p[1]; }
      { const uint4* p = (const uint4*)(wpU + kg); fbU.q[0] = p[0]; fbU.q[1] = p[1]; }
      { const uint4* p = (const uint4*)(wpO + kg); fbO.q[0] = p[0]; fbO.q[1] = p[1]; }
      // two A fragments (rows 0..15 and 16..31)
      FragAB fa0, fa1;
      const uint4* a0 = (const uint4*)&ldsA[lm * 128 + (kl >> 1) + 8 * half];
      const uint4* a1 = (const uint4*)&ldsA[(16 + lm) * 128 + (kl >> 1) + 8 * half];
      fa0.q[0] = a0[0]; fa0.q[1] = a0[1];
      fa1.q[0] = a1[0]; fa1.q[1] = a1[1];

      aF[0] = __builtin_amdgcn_wmma_f32_16x16x32_bf16(false, fa0.v, false, fbF.v, (short)0, aF[0], false, false);
      aF[1] = __builtin_amdgcn_wmma_f32_16x16x32_bf16(false, fa1.v, false, fbF.v, (short)0, aF[1], false, false);
      aI[0] = __builtin_amdgcn_wmma_f32_16x16x32_bf16(false, fa0.v, false, fbI.v, (short)0, aI[0], false, false);
      aI[1] = __builtin_amdgcn_wmma_f32_16x16x32_bf16(false, fa1.v, false, fbI.v, (short)0, aI[1], false, false);
      aU[0] = __builtin_amdgcn_wmma_f32_16x16x32_bf16(false, fa0.v, false, fbU.v, (short)0, aU[0], false, false);
      aU[1] = __builtin_amdgcn_wmma_f32_16x16x32_bf16(false, fa1.v, false, fbU.v, (short)0, aU[1], false, false);
      aO[0] = __builtin_amdgcn_wmma_f32_16x16x32_bf16(false, fa0.v, false, fbO.v, (short)0, aO[0], false, false);
      aO[1] = __builtin_amdgcn_wmma_f32_16x16x32_bf16(false, fa1.v, false, fbO.v, (short)0, aO[1], false, false);
    }
    __syncthreads();
  }

  // reduce q partials across the 4 column segments, finalize q_in
  {
    const int row = tid >> 2, j0 = (tid & 3) * 2;
#pragma unroll
    for (int jj = 0; jj < 2; ++jj) {
      const int j = j0 + jj;
      float s = 0.0f;
#pragma unroll
      for (int sg = 0; sg < 4; ++sg) s += qpartS[(row * 4 + sg) * NQ + j];
      qinS[row * NQ + j] = s + bq[j];
    }
  }
  __syncthreads();
  // cumprod-cos for 4 gates (128 threads: gate x row)
  {
    const int g = tid >> 5, row = tid & 31;
    const float* th = (g == 0) ? thf : (g == 1) ? thi : (g == 2) ? thu : tho;
    float cp = 1.0f;
#pragma unroll
    for (int j = 0; j < NQ; ++j) {
      cp *= __cosf(qinS[row * NQ + j] + th[j]);
      qcpS[(g * 32 + row) * NQ + j] = cp;
    }
  }
  __syncthreads();

  // epilogue: quantum mix, activations, cell update, write hidden state
  const float biasF = bF[h], biasI = bI[h], biasU = bU[h], biasO = bO[h], biasQ = bqh[h];
  float wq[NQ];
#pragma unroll
  for (int j = 0; j < NQ; ++j) wq[j] = Wqh[(size_t)j * HID + h];

#pragma unroll
  for (int m = 0; m < 2; ++m) {
#pragma unroll
    for (int r = 0; r < 8; ++r) {
      const int lrow = m * 16 + 8 * half + r;     // block-relative row
      const int brow = rowBase + lrow;            // global batch row
      float qf = 0.f, qi = 0.f, qu = 0.f, qo = 0.f;
#pragma unroll
      for (int j = 0; j < NQ; ++j) {
        const float w = wq[j];
        qf += qcpS[(0 * 32 + lrow) * NQ + j] * w;
        qi += qcpS[(1 * 32 + lrow) * NQ + j] * w;
        qu += qcpS[(2 * 32 + lrow) * NQ + j] * w;
        qo += qcpS[(3 * 32 + lrow) * NQ + j] * w;
      }
      const float fg = ALPHA * sigmf(aF[m][r] + biasF) + (1.0f - ALPHA) * sigmf(qf + biasQ);
      const float ig = ALPHA * sigmf(aI[m][r] + biasI) + (1.0f - ALPHA) * sigmf(qi + biasQ);
      const float gg = ALPHA * tanhf(aU[m][r] + biasU) + (1.0f - ALPHA) * tanhf(qu + biasQ);
      const float og = ALPHA * sigmf(aO[m][r] + biasO) + (1.0f - ALPHA) * sigmf(qo + biasQ);
      const size_t idx = (size_t)brow * HID + h;
      const float c1 = fg * cx[idx] + ig * gg;
      cx[idx]    = c1;
      out_t[idx] = og * tanhf(c1);
    }
  }
}

__global__ void qlstm_final(const float* __restrict__ outlast, const float* __restrict__ cxws,
                            float* __restrict__ dsth, float* __restrict__ dstc) {
  int i = blockIdx.x * blockDim.x + threadIdx.x;
  if (i < BATCH * HID) { dsth[i] = outlast[i]; dstc[i] = cxws[i]; }
}

extern "C" void kernel_launch(void* const* d_in, const int* in_sizes, int n_in,
                              void* d_out, int out_size, void* d_ws, size_t ws_size,
                              hipStream_t stream) {
  (void)in_sizes; (void)n_in; (void)out_size; (void)ws_size;
  const float* inputs = (const float*)d_in[0];
  const float* Wf  = (const float*)d_in[1];  const float* bf  = (const float*)d_in[2];
  const float* Wi  = (const float*)d_in[3];  const float* bi  = (const float*)d_in[4];
  const float* Wu  = (const float*)d_in[5];  const float* bu  = (const float*)d_in[6];
  const float* Wo  = (const float*)d_in[7];  const float* bo  = (const float*)d_in[8];
  const float* Wq  = (const float*)d_in[9];  const float* bq  = (const float*)d_in[10];
  const float* Wqh = (const float*)d_in[11]; const float* bqh = (const float*)d_in[12];
  const float* thf = (const float*)d_in[13]; const float* thi = (const float*)d_in[14];
  const float* thu = (const float*)d_in[15]; const float* tho = (const float*)d_in[16];
  float* out = (float*)d_out;

  char* ws = (char*)d_ws;
  __bf16* WT  = (__bf16*)(ws);              // 4*512*1024*2      = 4,194,304 B
  float*  cx  = (float*)(ws + 4194304);     // 512*512*4         = 1,048,576 B
  float*  hz  = (float*)(ws + 5242880);     // 512*512*4         = 1,048,576 B (zero hx)
  float*  WqT = (float*)(ws + 6291456);     // 8*1024*4          =    32,768 B

  qlstm_wprep<<<8192, 256, 0, stream>>>(Wf, Wi, Wu, Wo, WT);
  qlstm_wqprep<<<32, 256, 0, stream>>>(Wq, WqT);
  qlstm_zero<<<2048, 256, 0, stream>>>(cx, 2 * BATCH * HID);  // zeros cx + hz (contiguous)

  dim3 gblock(128);
  dim3 ggrid(BATCH / 32, HID / 64);
  for (int t = 0; t < T_STEPS; ++t) {
    const float* xt = inputs + (size_t)t * BATCH * DIM;
    const float* hx = (t == 0) ? hz : (out + (size_t)(t - 1) * BATCH * HID);
    qlstm_step<<<ggrid, gblock, 0, stream>>>(xt, hx, WT, WqT, bf, bi, bu, bo, Wqh, bqh,
                                             bq, thf, thi, thu, tho, cx,
                                             out + (size_t)t * BATCH * HID);
  }
  qlstm_final<<<1024, 256, 0, stream>>>(out + (size_t)(T_STEPS - 1) * BATCH * HID, cx,
                                        out + (size_t)T_STEPS * BATCH * HID,
                                        out + (size_t)T_STEPS * BATCH * HID + BATCH * HID);
}
